// MessagePassingNet_76931454206493
// MI455X (gfx1250) — matrix operand order repaired and checked by reading, not compiled, not run on hardware
//
#include <hip/hip_runtime.h>
#include <hip/hip_bf16.h>

// ---------------------------------------------------------------------------
// GraphSAGE + TopKPooling network for MI455X (gfx1250, wave32, WMMA)
// ---------------------------------------------------------------------------

typedef __attribute__((ext_vector_type(16))) _Float16 v16h;
typedef __attribute__((ext_vector_type(2)))  _Float16 v2h;
typedef __attribute__((ext_vector_type(8)))  float    v8f;

#define NB    512          // graphs
#define NPG   256          // nodes / graph (layer 1)
#define EPG   4096
#define NTOT  (NB * NPG)   // 131072
#define ETOT  (NB * EPG)   // 2097152
#define HF    128          // hidden == features
#define K1    205
#define K2    164
#define K3    132
#define M2    (NB * K1)    // 104960
#define M3    (NB * K2)    // 83968
#define M4    (NB * K3)    // 67584

// ---------------------------------------------------------------- fill utils
__global__ void fill_f32(float* p, float v, int n) {
    int i = blockIdx.x * blockDim.x + threadIdx.x;
    if (i < n) p[i] = v;
}
__global__ void fill_i32(int* p, int v, int n) {
    int i = blockIdx.x * blockDim.x + threadIdx.x;
    if (i < n) p[i] = v;
}

// copy input edges into workspace, valid = 1
__global__ void init_edges(const int* src_in, const int* dst_in,
                           int* src, int* dst, int* valid, int n) {
    int i = blockIdx.x * blockDim.x + threadIdx.x;
    if (i < n) { src[i] = src_in[i]; dst[i] = dst_in[i]; valid[i] = 1; }
}

// inv[m] = 1 / max(cnt[m], 1)  (full-precision divide, once per node)
__global__ void recip_cnt(const float* __restrict__ cnt, float* inv, int n) {
    int i = blockIdx.x * blockDim.x + threadIdx.x;
    if (i < n) inv[i] = 1.0f / fmaxf(cnt[i], 1.0f);
}

// ------------------------------------------------------------ weight prep
// Build [Wl ; Wr]^T (K=256 x N=128) in f16, pre-swizzled into WMMA B-fragment
// order:  WcS[kt*4096 + nt*512 + lane*16 + h]  holds element
//   k = kt*32 + ((lane>>4)<<4) + h ,  n = nt*16 + (lane&15)
// so each wave reads a B fragment as one contiguous 32-byte v16h load.
__global__ void build_wcat(const float* Wl, const float* Wr, _Float16* WcS) {
    int t = blockIdx.x * blockDim.x + threadIdx.x;
    if (t >= 256 * 128) return;
    int kt   = t >> 12;
    int nt   = (t >> 9) & 7;
    int lane = (t >> 4) & 31;
    int h    = t & 15;
    int k = kt * 32 + ((lane >> 4) << 4) + h;
    int n = nt * 16 + (lane & 15);
    float v = (k < 128) ? Wl[n * 128 + k] : Wr[n * 128 + (k - 128)];
    WcS[t] = (_Float16)v;
}

// wn = w / ||w||   (128 threads, 1 block)
__global__ void norm_w(const float* w, float* wn) {
    __shared__ float red[128];
    int t = threadIdx.x;
    float v = w[t];
    red[t] = v * v;
    __syncthreads();
    for (int s = 64; s > 0; s >>= 1) {
        if (t < s) red[t] += red[t + s];
        __syncthreads();
    }
    wn[t] = v * rsqrtf(red[0]);
}

// ------------------------------------------------------- segment sum (mean)
// one wave per edge; lane handles 4 features (float4 load + 4 f32 atomics)
__global__ __launch_bounds__(256)
void seg_sum(const float* __restrict__ x, const int* __restrict__ src,
             const int* __restrict__ dst, const int* __restrict__ valid,
             float* s, float* cnt, int nE) {
    int wave = threadIdx.x >> 5;
    int lane = threadIdx.x & 31;
    int e = blockIdx.x * 8 + wave;
    if (e >= nE || !valid[e]) return;
    int sn = src[e], dn = dst[e];
    float4 v = *(const float4*)(x + (size_t)sn * HF + lane * 4);
    float* d = s + (size_t)dn * HF + lane * 4;
    atomicAdd(d + 0, v.x);
    atomicAdd(d + 1, v.y);
    atomicAdd(d + 2, v.z);
    atomicAdd(d + 3, v.w);
    if (lane == 0) atomicAdd(&cnt[dn], 1.0f);
}

// ------------------------------------------------------- fused SAGE GEMM
// out[M,128] = relu( [mean | x] (M x 256) @ [Wl;Wr]^T (256 x 128) + bias )
// mean = s * inv (inv precomputed). f16 inputs, f32 WMMA accumulate.
// block = 256 threads (8 waves); block tile = 128(M) x 128(N); K-step 32.
// A tile staged in LDS in fragment order; B fragments loaded directly from
// the pre-swizzled (L2-resident) weight buffer as v16h (b128 pair) loads.
__global__ __launch_bounds__(256)
void sage_gemm(const float* __restrict__ s, const float* __restrict__ invc,
               const float* __restrict__ x, const _Float16* __restrict__ WcS,
               const float* __restrict__ bias, float* __restrict__ out, int M) {
    // AS[waveTile(8)][lane(32)][h(16)] : fragment-order A tile (8 KB)
    __shared__ __align__(32) _Float16 AS[8 * 512];

    int tid  = threadIdx.x;
    int wave = tid >> 5;
    int lane = tid & 31;
    int m0   = blockIdx.x * 128;

    v8f acc[8];
    v8f zv = {};
#pragma unroll
    for (int i = 0; i < 8; ++i) acc[i] = zv;

    for (int kt = 0; kt < 8; ++kt) {          // K = 256, step 32
        int kbase = kt * 32;                  // multiple of 32: whole tile is
        bool fromMean = (kbase < 128);        // on one side of the mean|x seam

        // ---- stage A tile: 2048 half-pairs, fragment-order LDS layout.
        // pair p -> halves AS[2p], AS[2p+1]; inverse fragment mapping:
        //   wt = (2p)>>9, ln = ((2p)>>4)&31, h0 = (2p)&15 (even)
        //   row r = wt*16 + (ln&15)
        //   kk0 = ((ln>>4)<<4) + h0   (even; kk1 = kk0+1)
#pragma unroll
        for (int j = 0; j < 8; ++j) {
            int p  = tid + j * 256;
            int hi = p << 1;
            int wt = hi >> 9;
            int ln = (hi >> 4) & 31;
            int h0 = hi & 15;
            int r  = wt * 16 + (ln & 15);
            int kk = ((ln >> 4) << 4) + h0;
            int m  = m0 + r;
            float v0, v1;
            if (fromMean) {
                float inv = invc[m];
                float2 t2 = *(const float2*)(s + (size_t)m * HF + kbase + kk);
                v0 = t2.x * inv;
                v1 = t2.y * inv;
            } else {
                float2 t2 = *(const float2*)(x + (size_t)m * HF + (kbase - 128) + kk);
                v0 = t2.x;
                v1 = t2.y;
            }
            if (j == 0 && kt + 1 < 8) {
                int kn = kbase + 32;
                const float* nsrc = (kn < 128) ? (s + (size_t)m * HF + kn + kk)
                                               : (x + (size_t)m * HF + (kn - 128) + kk);
                __builtin_prefetch(nsrc, 0, 3);   // locality 3 -> near (WGP) cache
            }
            v2h pk;
            pk[0] = (_Float16)v0;
            pk[1] = (_Float16)v1;
            *(v2h*)(&AS[hi]) = pk;
        }
        __syncthreads();

        // ---- A fragment: one contiguous 32B LDS read per wave
        v16h a = *(const v16h*)(&AS[wave * 512 + lane * 16]);

        // ---- 8 column tiles: contiguous 32B global B fragment + WMMA
        const _Float16* wb = WcS + (size_t)kt * 4096 + lane * 16;
#pragma unroll
        for (int nt = 0; nt < 8; ++nt) {
            v16h b = *(const v16h*)(wb + nt * 512);
            acc[nt] = __builtin_amdgcn_wmma_f32_16x16x32_f16(
                false, a, false, b, (short)0, acc[nt], false, false);
        }
        __syncthreads();
    }

    // ---- epilogue: bias + relu. C/D layout: vgpr r -> M = r + (lane>=16)*8
    int mbase = m0 + wave * 16 + ((lane >> 4) << 3);
    int nlow  = lane & 15;
#pragma unroll
    for (int nt = 0; nt < 8; ++nt) {
        int n = nt * 16 + nlow;
        float bv = bias[n];
#pragma unroll
        for (int r = 0; r < 8; ++r) {
            int m = mbase + r;
            float v = acc[nt][r] + bv;
            out[(size_t)m * HF + n] = fmaxf(v, 0.0f);
        }
    }
}

// ------------------------------------------------------- TopK pooling
// one block (256 thr) per graph: score = tanh(h . wn), bitonic sort desc,
// keep first K (ties broken by lower index, matching lax.top_k).
__global__ __launch_bounds__(256)
void topk_pool(const float* __restrict__ h, const float* __restrict__ wn,
               int n, int K, int* keep_idx, float* keep_val, int* newmap) {
    int g = blockIdx.x;
    int t = threadIdx.x;
    __shared__ float key[256];
    __shared__ int   idx[256];

    float sc = -__builtin_inff();
    if (t < n) {
        const float4* row = (const float4*)(h + (size_t)(g * n + t) * HF);
        const float4* wv  = (const float4*)wn;
        float acc = 0.0f;
        for (int f = 0; f < HF / 4; ++f) {
            float4 a = row[f], b = wv[f];
            acc += a.x * b.x + a.y * b.y + a.z * b.z + a.w * b.w;
        }
        sc = tanhf(acc);
    }
    key[t] = sc;
    idx[t] = t;
    __syncthreads();

    for (int ksz = 2; ksz <= 256; ksz <<= 1) {
        for (int j = ksz >> 1; j > 0; j >>= 1) {
            int ixj = t ^ j;
            if (ixj > t) {
                bool up = ((t & ksz) == 0);   // descending segments
                float ka = key[t], kb = key[ixj];
                int   ia = idx[t], ib = idx[ixj];
                bool aFirst = (ka > kb) || (ka == kb && ia < ib);
                if (up ? !aFirst : aFirst) {
                    key[t] = kb; key[ixj] = ka;
                    idx[t] = ib; idx[ixj] = ia;
                }
            }
            __syncthreads();
        }
    }
    if (t < K) {
        int oldg = g * n + idx[t];
        int newg = g * K + t;
        keep_idx[newg] = oldg;
        keep_val[newg] = key[t];
        newmap[oldg]   = newg;
    }
}

// out[j, f] = h[keep_idx[j], f] * keep_val[j]
__global__ void gather_scale(const float* __restrict__ h,
                             const int* __restrict__ keep_idx,
                             const float* __restrict__ keep_val,
                             float* out, int nNew) {
    int i = blockIdx.x * blockDim.x + threadIdx.x;
    if (i >= nNew * HF) return;
    int j = i >> 7, f = i & 127;
    out[i] = h[(size_t)keep_idx[j] * HF + f] * keep_val[j];
}

__global__ void edge_remap(const int* __restrict__ si, const int* __restrict__ di,
                           const int* __restrict__ vi, const int* __restrict__ newmap,
                           int* so, int* dn, int* vo, int nE) {
    int e = blockIdx.x * blockDim.x + threadIdx.x;
    if (e >= nE) return;
    int s2 = newmap[si[e]];
    int d2 = newmap[di[e]];
    int ok = (vi[e] != 0) && (s2 >= 0) && (d2 >= 0);
    so[e] = ok ? s2 : 0;
    dn[e] = ok ? d2 : 0;
    vo[e] = ok;
}

// z[g, 0:128] += max_j h ; z[g, 128:256] += mean_j h   (one block/graph)
__global__ __launch_bounds__(128)
void gpool_add(const float* __restrict__ h, int K, float* z) {
    int g = blockIdx.x;
    int f = threadIdx.x;
    const float* base = h + (size_t)g * K * HF + f;
    float mx = -__builtin_inff(), sm = 0.0f;
    for (int j = 0; j < K; ++j) {
        float v = base[(size_t)j * HF];
        mx = fmaxf(mx, v);
        sm += v;
    }
    z[g * 256 + f]       += mx;
    z[g * 256 + 128 + f] += sm * (1.0f / K);
}

// out[r,o] = (relu?)(in[r] . W[o] + b[o])
__global__ void linear(const float* __restrict__ in, const float* __restrict__ W,
                       const float* __restrict__ b, float* out,
                       int I, int O, int rows, int doRelu) {
    int t = blockIdx.x * blockDim.x + threadIdx.x;
    if (t >= rows * O) return;
    int r = t / O, o = t % O;
    const float* ir = in + (size_t)r * I;
    const float* wr = W + (size_t)o * I;
    float acc = b[o];
    for (int k = 0; k < I; ++k) acc += ir[k] * wr[k];
    out[t] = doRelu ? fmaxf(acc, 0.0f) : acc;
}

// ---------------------------------------------------------------------------
extern "C" void kernel_launch(void* const* d_in, const int* in_sizes, int n_in,
                              void* d_out, int out_size, void* d_ws, size_t ws_size,
                              hipStream_t stream) {
    const float* x    = (const float*)d_in[0];
    const int*   ei   = (const int*)d_in[1];      // [2, E]: src then dst
    // d_in[2] = batch (implicit in equal-sized blocks; unused)
    const float* W1l = (const float*)d_in[3];
    const float* b1  = (const float*)d_in[4];
    const float* W1r = (const float*)d_in[5];
    const float* W2l = (const float*)d_in[6];
    const float* b2  = (const float*)d_in[7];
    const float* W2r = (const float*)d_in[8];
    const float* W3l = (const float*)d_in[9];
    const float* b3  = (const float*)d_in[10];
    const float* W3r = (const float*)d_in[11];
    const float* p1w = (const float*)d_in[12];
    const float* p2w = (const float*)d_in[13];
    const float* p3w = (const float*)d_in[14];
    const float* L1W = (const float*)d_in[15];
    const float* L1b = (const float*)d_in[16];
    const float* L2W = (const float*)d_in[17];
    const float* L2b = (const float*)d_in[18];
    const float* L3W = (const float*)d_in[19];
    const float* L3b = (const float*)d_in[20];
    float* out = (float*)d_out;

    const int* src_in = ei;
    const int* dst_in = ei + ETOT;

    // ---------------- workspace carve (deterministic layout) ----------------
    char* ws = (char*)d_ws;
    size_t off = 0;
    auto take = [&](size_t bytes) -> char* {
        char* p = ws + off;
        off = (off + bytes + 255) & ~(size_t)255;
        return p;
    };
    float*     S    = (float*)take((size_t)NTOT * HF * 4);   // segment sums
    float*     CNT  = (float*)take((size_t)NTOT * 4);
    float*     INV  = (float*)take((size_t)NTOT * 4);
    float*     buf0 = (float*)take((size_t)NTOT * HF * 4);   // h1 / h2 / h3
    float*     buf1 = (float*)take((size_t)NTOT * HF * 4);   // h1k / h2k / h3k
    int*       srcA = (int*)take((size_t)ETOT * 4);
    int*       dstA = (int*)take((size_t)ETOT * 4);
    int*       valA = (int*)take((size_t)ETOT * 4);
    int*       srcB = (int*)take((size_t)ETOT * 4);
    int*       dstB = (int*)take((size_t)ETOT * 4);
    int*       valB = (int*)take((size_t)ETOT * 4);
    _Float16*  Wc1  = (_Float16*)take((size_t)256 * 128 * 2);
    _Float16*  Wc2  = (_Float16*)take((size_t)256 * 128 * 2);
    _Float16*  Wc3  = (_Float16*)take((size_t)256 * 128 * 2);
    float*     WN1  = (float*)take(128 * 4);
    float*     WN2  = (float*)take(128 * 4);
    float*     WN3  = (float*)take(128 * 4);
    float*     Z    = (float*)take((size_t)NB * 256 * 4);
    float*     T1   = (float*)take((size_t)NB * 128 * 4);
    float*     T2   = (float*)take((size_t)NB * 64 * 4);
    int*       keepI = (int*)take((size_t)M2 * 4);
    float*     keepV = (float*)take((size_t)M2 * 4);
    int*       nmap  = (int*)take((size_t)NTOT * 4);

    auto blocks = [](int n) { return (n + 255) / 256; };

    // ---------------- prep ----------------
    build_wcat<<<blocks(256 * 128), 256, 0, stream>>>(W1l, W1r, Wc1);
    build_wcat<<<blocks(256 * 128), 256, 0, stream>>>(W2l, W2r, Wc2);
    build_wcat<<<blocks(256 * 128), 256, 0, stream>>>(W3l, W3r, Wc3);
    norm_w<<<1, 128, 0, stream>>>(p1w, WN1);
    norm_w<<<1, 128, 0, stream>>>(p2w, WN2);
    norm_w<<<1, 128, 0, stream>>>(p3w, WN3);
    init_edges<<<blocks(ETOT), 256, 0, stream>>>(src_in, dst_in, srcA, dstA, valA, ETOT);
    fill_f32<<<blocks(NB * 256), 256, 0, stream>>>(Z, 0.0f, NB * 256);

    // ---------------- layer 1 (N = 131072, n/graph = 256 -> K1 = 205) ------
    fill_f32<<<blocks(NTOT * HF), 256, 0, stream>>>(S, 0.0f, NTOT * HF);
    fill_f32<<<blocks(NTOT), 256, 0, stream>>>(CNT, 0.0f, NTOT);
    seg_sum<<<ETOT / 8, 256, 0, stream>>>(x, srcA, dstA, valA, S, CNT, ETOT);
    recip_cnt<<<blocks(NTOT), 256, 0, stream>>>(CNT, INV, NTOT);
    sage_gemm<<<NTOT / 128, 256, 0, stream>>>(S, INV, x, Wc1, b1, buf0, NTOT);
    fill_i32<<<blocks(NTOT), 256, 0, stream>>>(nmap, -1, NTOT);
    topk_pool<<<NB, 256, 0, stream>>>(buf0, WN1, NPG, K1, keepI, keepV, nmap);
    gather_scale<<<blocks(M2 * HF), 256, 0, stream>>>(buf0, keepI, keepV, buf1, M2);
    edge_remap<<<blocks(ETOT), 256, 0, stream>>>(srcA, dstA, valA, nmap, srcB, dstB, valB, ETOT);
    gpool_add<<<NB, 128, 0, stream>>>(buf1, K1, Z);

    // ---------------- layer 2 (M2 = 104960, n/graph = 205 -> K2 = 164) -----
    fill_f32<<<blocks(M2 * HF), 256, 0, stream>>>(S, 0.0f, M2 * HF);
    fill_f32<<<blocks(M2), 256, 0, stream>>>(CNT, 0.0f, M2);
    seg_sum<<<ETOT / 8, 256, 0, stream>>>(buf1, srcB, dstB, valB, S, CNT, ETOT);
    recip_cnt<<<blocks(M2), 256, 0, stream>>>(CNT, INV, M2);
    sage_gemm<<<M2 / 128, 256, 0, stream>>>(S, INV, buf1, Wc2, b2, buf0, M2);
    fill_i32<<<blocks(M2), 256, 0, stream>>>(nmap, -1, M2);
    topk_pool<<<NB, 256, 0, stream>>>(buf0, WN2, K1, K2, keepI, keepV, nmap);
    gather_scale<<<blocks(M3 * HF), 256, 0, stream>>>(buf0, keepI, keepV, buf1, M3);
    edge_remap<<<blocks(ETOT), 256, 0, stream>>>(srcB, dstB, valB, nmap, srcA, dstA, valA, ETOT);
    gpool_add<<<NB, 128, 0, stream>>>(buf1, K2, Z);

    // ---------------- layer 3 (M3 = 83968, n/graph = 164 -> K3 = 132) ------
    fill_f32<<<blocks(M3 * HF), 256, 0, stream>>>(S, 0.0f, M3 * HF);
    fill_f32<<<blocks(M3), 256, 0, stream>>>(CNT, 0.0f, M3);
    seg_sum<<<ETOT / 8, 256, 0, stream>>>(buf1, srcA, dstA, valA, S, CNT, ETOT);
    recip_cnt<<<blocks(M3), 256, 0, stream>>>(CNT, INV, M3);
    sage_gemm<<<M3 / 128, 256, 0, stream>>>(S, INV, buf1, Wc3, b3, buf0, M3);
    fill_i32<<<blocks(M3), 256, 0, stream>>>(nmap, -1, M3);
    topk_pool<<<NB, 256, 0, stream>>>(buf0, WN3, K2, K3, keepI, keepV, nmap);
    gather_scale<<<blocks(M4 * HF), 256, 0, stream>>>(buf0, keepI, keepV, buf1, M4);
    gpool_add<<<NB, 128, 0, stream>>>(buf1, K3, Z);

    // ---------------- MLP head ----------------
    linear<<<blocks(NB * 128), 256, 0, stream>>>(Z, L1W, L1b, T1, 256, 128, NB, 1);
    linear<<<blocks(NB * 64), 256, 0, stream>>>(T1, L2W, L2b, T2, 128, 64, NB, 1);
    linear<<<blocks(NB * 8), 256, 0, stream>>>(T2, L3W, L3b, out, 64, 8, NB, 0);
}
